// SelfAttentionLayer_12008728560272
// MI455X (gfx1250) — matrix operand order repaired and checked by reading, not compiled, not run on hardware
//
#include <hip/hip_runtime.h>
#include <stdint.h>

typedef __attribute__((ext_vector_type(16))) _Float16 v16h;
typedef __attribute__((ext_vector_type(8)))  float    v8f;

#define DEV_INLINE __device__ __forceinline__

constexpr int kB  = 4;
constexpr int kS  = 2048;
constexpr int kDM = 768;
constexpr int kH  = 12;
constexpr int kHD = 64;
constexpr int kM  = kB * kS;     // 8192 token rows
constexpr int kN3 = 3 * kDM;     // 2304
constexpr float kInvHeadZ = 0.125f;  // 1/sqrt(64)

// ---------------------------------------------------------------------------
// WMMA wrapper: D = A(16x32 f16) * B(32x16 f16) + C(f32)
// ---------------------------------------------------------------------------
DEV_INLINE v8f wmma16x16x32(v16h a, v16h b, v8f c) {
  return __builtin_amdgcn_wmma_f32_16x16x32_f16(
      /*neg_a=*/false, a, /*neg_b=*/false, b,
      /*c_mod=*/(short)0, c, /*reuse_a=*/false, /*reuse_b=*/false);
}

// ---------------------------------------------------------------------------
// Fragment loaders per CDNA5 ISA layouts (wave32).
// A 16x32 f16: lane L -> row m=L&15, half=L>>4; VGPR j<4: K=half*8+2j(+1);
//              VGPR j>=4: K=16+half*8+2(j-4)(+1).
// ---------------------------------------------------------------------------
DEV_INLINE v16h load_a_f16(const _Float16* A, int lda) {
  const int lane = threadIdx.x & 31;
  const int m = lane & 15, ph = lane >> 4;
  const _Float16* row = A + (size_t)m * lda + ph * 8;
  v16h a;
#pragma unroll
  for (int j = 0; j < 4; ++j) {
    a[2 * j]         = row[2 * j];
    a[2 * j + 1]     = row[2 * j + 1];
    a[8 + 2 * j]     = row[16 + 2 * j];
    a[8 + 2 * j + 1] = row[16 + 2 * j + 1];
  }
  return a;
}

// Same A layout but converting from f32 source on the fly.
DEV_INLINE v16h load_a_f32cvt(const float* A, int lda) {
  const int lane = threadIdx.x & 31;
  const int m = lane & 15, ph = lane >> 4;
  const float* row = A + (size_t)m * lda + ph * 8;
  v16h a;
#pragma unroll
  for (int j = 0; j < 4; ++j) {
    a[2 * j]         = (_Float16)row[2 * j];
    a[2 * j + 1]     = (_Float16)row[2 * j + 1];
    a[8 + 2 * j]     = (_Float16)row[16 + 2 * j];
    a[8 + 2 * j + 1] = (_Float16)row[16 + 2 * j + 1];
  }
  return a;
}

// A layout but source elements are f16 payloads stored every 2 f16 slots
// (f16 packed in the low half of 4-byte LDS cells). rowElems = elements/row.
DEV_INLINE v16h load_a_strided2(const _Float16* P, int rowElems) {
  const int lane = threadIdx.x & 31;
  const int m = lane & 15, ph = lane >> 4;
  const _Float16* row = P + ((size_t)m * rowElems + ph * 8) * 2;
  v16h a;
#pragma unroll
  for (int j = 0; j < 4; ++j) {
    a[2 * j]         = row[(2 * j) * 2];
    a[2 * j + 1]     = row[(2 * j + 1) * 2];
    a[8 + 2 * j]     = row[(16 + 2 * j) * 2];
    a[8 + 2 * j + 1] = row[(16 + 2 * j + 1) * 2];
  }
  return a;
}

// B 32x16 f16: lane L -> col n=L&15, half=L>>4; VGPR j<4: K=half*8+2j(+1);
//              VGPR j>=4: K=16+half*8+2(j-4)(+1).
// Source stored transposed [N,K]: value(k,n) = BT[n*ldb + k] (contiguous K).
DEV_INLINE v16h load_b_t(const _Float16* BT, int ldb) {
  const int lane = threadIdx.x & 31;
  const int n = lane & 15, ph = lane >> 4;
  const _Float16* row = BT + (size_t)n * ldb + ph * 8;
  v16h b;
#pragma unroll
  for (int j = 0; j < 8; ++j) {
    b[j]     = row[j];
    b[8 + j] = row[16 + j];
  }
  return b;
}

// B source stored natural row-major [K,N]: value(k,n) = Bm[k*ldb + n].
DEV_INLINE v16h load_b_k(const _Float16* Bm, int ldb) {
  const int lane = threadIdx.x & 31;
  const int n = lane & 15, ph = lane >> 4;
  const _Float16* p = Bm + (size_t)(ph * 8) * ldb + n;
  v16h b;
#pragma unroll
  for (int j = 0; j < 8; ++j) {
    b[j]     = p[(size_t)j * ldb];
    b[8 + j] = p[(size_t)(16 + j) * ldb];
  }
  return b;
}

// ---------------------------------------------------------------------------
// Async copy of 16 bytes global -> LDS (per lane), tracked by ASYNCcnt.
// ldsoff = raw LDS byte offset (flat-address low 32 bits per ISA aperture map).
// ---------------------------------------------------------------------------
DEV_INLINE void async_copy_b128(unsigned ldsoff, const void* gptr) {
  unsigned long long ga = (unsigned long long)(uintptr_t)gptr;
  asm volatile("global_load_async_to_lds_b128 %0, %1, off"
               :
               : "v"(ldsoff), "v"(ga)
               : "memory");
}

DEV_INLINE void wait_async_le2() {
  asm volatile("s_wait_asynccnt 0x2" ::: "memory");
}
DEV_INLINE void wait_async_le0() {
  asm volatile("s_wait_asynccnt 0x0" ::: "memory");
}

// ---------------------------------------------------------------------------
// Weight transpose + f32->f16 convert: src[K,N] -> dst[N,K]
// ---------------------------------------------------------------------------
__global__ __launch_bounds__(256) void transpose_cvt_kernel(
    const float* __restrict__ src, _Float16* __restrict__ dst, int K, int N) {
  size_t gid = (size_t)blockIdx.x * 256 + threadIdx.x;
  if (gid >= (size_t)K * N) return;
  int k = (int)(gid / N);
  int n = (int)(gid % N);
  dst[(size_t)n * K + k] = (_Float16)src[gid];
}

// ---------------------------------------------------------------------------
// Generic WMMA GEMM: C = A[M,K] @ BT[N,K]^T + bias (+epilogue)
// MODE 0: A f32 (convert), store f16             (QKV)
// MODE 1: A f16, store f32 of (acc+bias+res)     (FFN1, res=base)
// MODE 2: A f16, store f32 of relu(acc+bias)+res (FFN2, res=hidden)
//
// Block = 256 threads = 8 waves stacked in M (block tile 128 x 64).
// K-step = 64: the shared 64x64 B slab is staged into LDS with
// double-buffered global_load_async_to_lds_b128 (2 issues/thread/step, so
// ASYNCcnt <= 4 outstanding), overlapping the copy of step i+1 with the
// 8 WMMAs of step i. Only 2 workgroup barriers per 8 WMMAs/wave.
// Next A strip is prefetched (global_prefetch_b8).
// ---------------------------------------------------------------------------
template <int MODE>
__global__ __launch_bounds__(256) void gemm_kernel(
    const void* __restrict__ Av, const _Float16* __restrict__ BT,
    const float* __restrict__ bias, const float* __restrict__ res,
    void* __restrict__ Cv, int M, int N, int K) {
  __shared__ _Float16 bstage[2][64 * 64];  // 2 x 8KB double buffer

  const int tid = threadIdx.x;
  const int wave = tid >> 5;
  const int lane = tid & 31;
  const int cn = lane & 15, ph = lane >> 4;
  const int row0 = blockIdx.x * 128 + wave * 16;
  const int col0 = blockIdx.y * 64;

  // Cooperative B-slab copy: 64 cols x 64 K f16 = 512 chunks of 16B;
  // each of 256 threads issues 2 async copies (cols c_l and c_l+32).
  const int c_l = tid >> 3;  // 0..31
  const int c_q = tid & 7;   // 16B chunk within the 64-f16 K row
  const _Float16* bsrcA = BT + (size_t)(col0 + c_l) * K + 8 * c_q;
  const _Float16* bsrcB = BT + (size_t)(col0 + 32 + c_l) * K + 8 * c_q;
  const unsigned dA0 = (unsigned)(uintptr_t)(&bstage[0][c_l * 64 + 8 * c_q]);
  const unsigned dB0 =
      (unsigned)(uintptr_t)(&bstage[0][(32 + c_l) * 64 + 8 * c_q]);
  const unsigned dA1 = (unsigned)(uintptr_t)(&bstage[1][c_l * 64 + 8 * c_q]);
  const unsigned dB1 =
      (unsigned)(uintptr_t)(&bstage[1][(32 + c_l) * 64 + 8 * c_q]);

  auto issue_tile = [&](int buf, int kk) {
    if (buf == 0) {
      async_copy_b128(dA0, bsrcA + kk);
      async_copy_b128(dB0, bsrcB + kk);
    } else {
      async_copy_b128(dA1, bsrcA + kk);
      async_copy_b128(dB1, bsrcB + kk);
    }
  };

  v8f acc[4];
#pragma unroll
  for (int t = 0; t < 4; ++t)
#pragma unroll
    for (int j = 0; j < 8; ++j) acc[t][j] = 0.0f;

  // Prologue: start copy of first B slab into buffer 0.
  issue_tile(0, 0);

  const int nsteps = K >> 6;  // K/64 (K divisible by 64 here)
  for (int i = 0; i < nsteps; ++i) {
    const int k0 = i << 6;
    const int cur = i & 1;

    if (i + 1 < nsteps) {
      issue_tile(cur ^ 1, k0 + 64);  // kick off next slab
      wait_async_le2();              // current slab (in-order) is complete
    } else {
      wait_async_le0();
    }
    __syncthreads();  // staged B visible to all waves

    // Prefetch next A strip while compute proceeds.
    if (i + 1 < nsteps) {
      if (MODE == 0)
        __builtin_prefetch((const float*)Av + (size_t)(row0 + (lane & 15)) * K +
                               (k0 + 64),
                           0, 0);
      else
        __builtin_prefetch((const _Float16*)Av +
                               (size_t)(row0 + (lane & 15)) * K + (k0 + 64),
                           0, 0);
    }

    v16h a0, a1;
    if (MODE == 0) {
      a0 = load_a_f32cvt((const float*)Av + (size_t)row0 * K + k0, K);
      a1 = load_a_f32cvt((const float*)Av + (size_t)row0 * K + k0 + 32, K);
    } else {
      a0 = load_a_f16((const _Float16*)Av + (size_t)row0 * K + k0, K);
      a1 = load_a_f16((const _Float16*)Av + (size_t)row0 * K + k0 + 32, K);
    }

    const _Float16* bt = bstage[cur];  // [64][64] local [N,K] layout
#pragma unroll
    for (int t = 0; t < 4; ++t) {
      v16h b0 = load_b_t(bt + (size_t)(16 * t) * 64, 64);
      acc[t] = wmma16x16x32(a0, b0, acc[t]);
    }
#pragma unroll
    for (int t = 0; t < 4; ++t) {
      v16h b1 = load_b_t(bt + (size_t)(16 * t) * 64 + 32, 64);
      acc[t] = wmma16x16x32(a1, b1, acc[t]);
    }
    __syncthreads();  // everyone done reading before buffer is overwritten
  }

#pragma unroll
  for (int t = 0; t < 4; ++t) {
#pragma unroll
    for (int j = 0; j < 8; ++j) {
      int row = row0 + j + 8 * ph;
      int col = col0 + 16 * t + cn;
      size_t idx = (size_t)row * N + col;
      float v = acc[t][j] + bias[col];
      if (MODE == 0) {
        ((_Float16*)Cv)[idx] = (_Float16)v;
      } else if (MODE == 1) {
        ((float*)Cv)[idx] = v + res[idx];
      } else {
        v = v > 0.0f ? v : 0.0f;
        ((float*)Cv)[idx] = v + res[idx];
      }
    }
  }
}

// ---------------------------------------------------------------------------
// Attention: one block per (b, h, 16-query tile). 256 threads = 8 waves.
// Phase 1: scores 16x2048 f32 into LDS via WMMA (Q frags loaded once).
// Phase 2: row softmax; probs rewritten in-place as f16 in low half of cells.
// Phase 3: waves 0..3 each own a 16-wide slice of HD=64; PV via WMMA.
// LDS = 16*2048*4 = 128 KB (CDNA5 allows up to 320 KB per workgroup).
// ---------------------------------------------------------------------------
__global__ __launch_bounds__(256) void attn_kernel(
    const _Float16* __restrict__ qkv, _Float16* __restrict__ av) {
  extern __shared__ char smem[];
  float* sc = (float*)smem;  // [16][kS]

  const int qt = blockIdx.x & 127;            // kS/16 = 128
  const int h  = (blockIdx.x >> 7) % kH;
  const int b  = blockIdx.x / (128 * kH);
  const int tid = threadIdx.x;
  const int wave = tid >> 5, lane = tid & 31;
  const int cn = lane & 15, ph = lane >> 4;

  const _Float16* qb = qkv + (size_t)b * kS * kN3 + h * kHD;
  const _Float16* kb = qb + kDM;
  const _Float16* vb = qb + 2 * kDM;

  // Phase 1: scores = (Q @ K^T) / sqrt(HD)
  const _Float16* Aq = qb + (size_t)(qt * 16) * kN3;
  v16h a0 = load_a_f16(Aq, kN3);       // K dims 0..31
  v16h a1 = load_a_f16(Aq + 32, kN3);  // K dims 32..63
  for (int kt = wave; kt < kS / 16; kt += 8) {
    const _Float16* Bk = kb + (size_t)(kt * 16) * kN3;  // value(k,n)=Bk[n*kN3+k]
    v8f c;
#pragma unroll
    for (int j = 0; j < 8; ++j) c[j] = 0.0f;
    c = wmma16x16x32(a0, load_b_t(Bk, kN3), c);
    c = wmma16x16x32(a1, load_b_t(Bk + 32, kN3), c);
#pragma unroll
    for (int j = 0; j < 8; ++j)
      sc[(j + 8 * ph) * kS + kt * 16 + cn] = c[j] * kInvHeadZ;
  }
  __syncthreads();

  // Phase 2: softmax per row; 16 threads per row, shfl reductions width 16.
  {
    const int r = tid >> 4, t = tid & 15;
    float mx = -3.0e38f;
    for (int i = t; i < kS; i += 16) mx = fmaxf(mx, sc[r * kS + i]);
#pragma unroll
    for (int off = 8; off > 0; off >>= 1) mx = fmaxf(mx, __shfl_xor(mx, off, 16));
    float sm = 0.0f;
    for (int i = t; i < kS; i += 16) sm += __expf(sc[r * kS + i] - mx);
#pragma unroll
    for (int off = 8; off > 0; off >>= 1) sm += __shfl_xor(sm, off, 16);
    float inv = 1.0f / sm;
    for (int i = t; i < kS; i += 16) {
      float p = __expf(sc[r * kS + i] - mx) * inv;
      ((_Float16*)&sc[r * kS + i])[0] = (_Float16)p;  // in-place f16 payload
    }
  }
  __syncthreads();

  // Phase 3: O = P @ V ; wave w handles head-dim slice [16w, 16w+16)
  if (wave < 4) {
    v8f acc;
#pragma unroll
    for (int j = 0; j < 8; ++j) acc[j] = 0.0f;
    const _Float16* P = (const _Float16*)smem;
    for (int kt = 0; kt < kS / 32; ++kt) {
      v16h ap = load_a_strided2(P + (size_t)(kt * 32) * 2, kS);
      v16h bv = load_b_k(vb + (size_t)(kt * 32) * kN3 + wave * 16, kN3);
      acc = wmma16x16x32(ap, bv, acc);
    }
#pragma unroll
    for (int j = 0; j < 8; ++j) {
      int row = b * kS + qt * 16 + j + 8 * ph;
      av[(size_t)row * kDM + h * kHD + wave * 16 + cn] = (_Float16)acc[j];
    }
  }
}

// ---------------------------------------------------------------------------
// Row normalization: (x-mean)/std * scale + bias over last dim (768).
// One block (256 threads) per row. Optionally emits an f16 copy for next GEMM.
// ---------------------------------------------------------------------------
template <bool WRITE_H>
__global__ __launch_bounds__(256) void norm_kernel(
    const float* __restrict__ x, const float* __restrict__ scalep,
    const float* __restrict__ biasp, float* __restrict__ out,
    _Float16* __restrict__ outh) {
  __shared__ float rs[8], rq[8];
  const int row = blockIdx.x;
  const int tid = threadIdx.x;
  const float* xr = x + (size_t)row * kDM;

  float s = 0.0f, q = 0.0f;
  for (int i = tid; i < kDM; i += 256) {
    float v = xr[i];
    s += v;
    q += v * v;
  }
#pragma unroll
  for (int off = 16; off > 0; off >>= 1) {
    s += __shfl_xor(s, off, 32);
    q += __shfl_xor(q, off, 32);
  }
  const int wave = tid >> 5, lane = tid & 31;
  if (lane == 0) { rs[wave] = s; rq[wave] = q; }
  __syncthreads();
  if (tid == 0) {
    float ts = 0.0f, tq = 0.0f;
#pragma unroll
    for (int i = 0; i < 8; ++i) { ts += rs[i]; tq += rq[i]; }
    rs[0] = ts;
    rq[0] = tq;
  }
  __syncthreads();
  float mean = rs[0] * (1.0f / kDM);
  float var  = rq[0] * (1.0f / kDM) - mean * mean;
  float rstd = 1.0f / sqrtf(var);
  float scl = scalep[0], bia = biasp[0];
  float* orow = out + (size_t)row * kDM;
  for (int i = tid; i < kDM; i += 256) {
    float y = (xr[i] - mean) * rstd * scl + bia;
    orow[i] = y;
    if (WRITE_H) outh[(size_t)row * kDM + i] = (_Float16)y;
  }
}

// ---------------------------------------------------------------------------
// Host-side orchestration
// ---------------------------------------------------------------------------
extern "C" void kernel_launch(void* const* d_in, const int* in_sizes, int n_in,
                              void* d_out, int out_size, void* d_ws, size_t ws_size,
                              hipStream_t stream) {
  (void)in_sizes; (void)n_in; (void)out_size; (void)ws_size;
  const float* base = (const float*)d_in[0];
  const float* Wqkv = (const float*)d_in[1];
  const float* bqkv = (const float*)d_in[2];
  const float* W1   = (const float*)d_in[3];
  const float* b1   = (const float*)d_in[4];
  const float* W2   = (const float*)d_in[5];
  const float* b2   = (const float*)d_in[6];
  const float* n1s  = (const float*)d_in[7];
  const float* n1b  = (const float*)d_in[8];
  const float* n2s  = (const float*)d_in[9];
  const float* n2b  = (const float*)d_in[10];

  char* ws = (char*)d_ws;
  // Workspace layout (bytes):
  _Float16* qkv_h    = (_Float16*)(ws + 0);          // 8192*2304*2 = 37,748,736
  _Float16* wqkvT    = (_Float16*)(ws + 37748736);   // 2304*768*2  =  3,538,944
  _Float16* w1T      = (_Float16*)(ws + 41287680);   // 768*768*2   =  1,179,648
  _Float16* w2T      = (_Float16*)(ws + 42467328);   // 768*768*2   =  1,179,648
  _Float16* av_h     = (_Float16*)(ws + 43646976);   // 8192*768*2  = 12,582,912
  float*    pre      = (float*)   (ws + 56229888);   // 8192*768*4  = 25,165,824
  float*    hidden   = (float*)   (ws + 81395712);   // 8192*768*4  = 25,165,824
  _Float16* hidden_h = (_Float16*)(ws + 106561536);  // 8192*768*2  = 12,582,912

  // 1) weight transpose+convert to f16 [N,K]
  transpose_cvt_kernel<<<(kDM * kN3 + 255) / 256, 256, 0, stream>>>(Wqkv, wqkvT, kDM, kN3);
  transpose_cvt_kernel<<<(kDM * kDM + 255) / 256, 256, 0, stream>>>(W1, w1T, kDM, kDM);
  transpose_cvt_kernel<<<(kDM * kDM + 255) / 256, 256, 0, stream>>>(W2, w2T, kDM, kDM);

  // 2) QKV = base @ Wqkv + bqkv   (f32 A converted, f16 output)
  gemm_kernel<0><<<dim3(kM / 128, kN3 / 64), 256, 0, stream>>>(
      (const void*)base, wqkvT, bqkv, nullptr, (void*)qkv_h, kM, kN3, kDM);

  // 3) attention -> av (f16)
  attn_kernel<<<kB * kH * (kS / 16), 256, 16 * kS * sizeof(float), stream>>>(qkv_h, av_h);

  // 4) pre1 = av @ W1 + b1 + base  (f32)
  gemm_kernel<1><<<dim3(kM / 128, kDM / 64), 256, 0, stream>>>(
      (const void*)av_h, w1T, b1, base, (void*)pre, kM, kDM, kDM);

  // 5) hidden = norm1(pre1)  (f32 + f16 copy)
  norm_kernel<true><<<kM, 256, 0, stream>>>(pre, n1s, n1b, hidden, hidden_h);

  // 6) pre2 = relu(hidden @ W2 + b2) + hidden  (f32)
  gemm_kernel<2><<<dim3(kM / 128, kDM / 64), 256, 0, stream>>>(
      (const void*)hidden_h, w2T, b2, hidden, (void*)pre, kM, kDM, kDM);

  // 7) out = norm2(pre2)
  norm_kernel<false><<<kM, 256, 0, stream>>>(pre, n2s, n2b, (float*)d_out, nullptr);
}
